// BiLSTMTagger_61022895341989
// MI455X (gfx1250) — compile-verified, matrix operand order checked by reading, not run on hardware
//
#include <hip/hip_runtime.h>

// BiLSTM tagger for gfx1250 (MI455X): WMMA f16 GEMMs + persistent LSTM scan
// with TDM (tensor_load_to_lds) double-buffered prefetch of the pre-activation
// slice. VOCAB=50000, TAGS=50, EMB=128, HID=256, B=128, T=256

typedef __attribute__((ext_vector_type(16))) _Float16 v16h;
typedef __attribute__((ext_vector_type(8)))  float    v8f;
typedef unsigned int v4u __attribute__((ext_vector_type(4)));
typedef int          v8i __attribute__((ext_vector_type(8)));
typedef int          v4i __attribute__((ext_vector_type(4)));

#define EMB_  128
#define HID_  256
#define B_    128
#define T_    256
#define G4_   1024   // 4*HID
#define TAGS_ 50
#define MROWS (T_ * B_)   // 32768
#define PRE_TILE_ELEMS (16 * G4_)      // 16384 f32 = 64KB per timestep slice

// ---------------------------------------------------------------------------
// Pack a row-major f32 weight W[N,K] into the wave32 WMMA B-fragment layout
// (16-bit A/B layout, ISA 7.12.2): tile (nt,kt) -> 32 lanes x 16 halfs,
// lane holds column n = nt*16 + (lane&15), half = lane>>4,
// VGPR v holds K pair kb(v,half) = (v<4 ? 2v : 16+2(v-4)) + 8*half.
// Zero-pads columns n >= N (used for TAGS=50 -> 64).
// ---------------------------------------------------------------------------
__global__ __launch_bounds__(256)
void pack_w16(const float* __restrict__ W, _Float16* __restrict__ P,
              int N, int K, int total) {
  int idx = blockIdx.x * blockDim.x + threadIdx.x;
  if (idx >= total) return;
  int e    = idx & 15;
  int lane = (idx >> 4) & 31;
  int tile = idx >> 9;
  int KT   = K >> 5;
  int kt   = tile % KT;
  int nt   = tile / KT;
  int n    = nt * 16 + (lane & 15);
  int half = lane >> 4;
  int v    = e >> 1;
  int kb   = (v < 4 ? 2 * v : 16 + 2 * (v - 4)) + 8 * half;
  int k    = kt * 32 + kb + (e & 1);
  float val = (n < N) ? W[(size_t)n * K + k] : 0.0f;
  P[idx] = (_Float16)val;
}

// ---------------------------------------------------------------------------
// Embedding gather: xs[t*B+b][e] = f16(emb[sentence[b][t]][e])
// ---------------------------------------------------------------------------
__global__ __launch_bounds__(256)
void embed_gather(const int* __restrict__ sent, const float* __restrict__ emb,
                  _Float16* __restrict__ xs) {
  int idx = blockIdx.x * blockDim.x + threadIdx.x;
  if (idx >= MROWS * EMB_) return;
  int e  = idx % EMB_;
  int tb = idx / EMB_;
  int b  = tb % B_;
  int t  = tb / B_;
  int tok = sent[b * T_ + t];
  xs[idx] = (_Float16)emb[(size_t)tok * EMB_ + e];
}

// Build the per-lane A fragment (16x32 f16, ISA 7.12.2) from a row-major
// f16 matrix row pointer (row = m0 + (lane&15), base col = kt*32).
__device__ inline v16h load_a_frag(const _Float16* __restrict__ arow, int half) {
  v16h a;
#pragma unroll
  for (int v = 0; v < 8; ++v) {
    int kb = (v < 4 ? 2 * v : 16 + 2 * (v - 4)) + 8 * half;
    a[2 * v]     = arow[kb];
    a[2 * v + 1] = arow[kb + 1];
  }
  return a;
}

// ---------------------------------------------------------------------------
// TDM: DMA one contiguous 64KB pre[t] slice (16 rows x 1024 f32) to LDS.
// D# per CDNA5 ISA: group0 = {count, lds_addr,
// global_addr[31:0], global_addr[56:32] | type=2<<30}; group1 packs
// data_size=4B, tensor_dim0=tile_dim0=16384, tensor_dim1=tile_dim1=1.
// Groups 2/3 (+extended) unused for a 1-row tile -> zero-filled.
// Tracked by TENSORcnt; waited with s_wait_tensorcnt.
// ---------------------------------------------------------------------------
__device__ inline void tdm_load_pre(const float* __restrict__ gsrc,
                                    unsigned ldsOff) {
  unsigned long long ga = (unsigned long long)(uintptr_t)gsrc;
  v4u g0;
  g0.x = 1u;                                         // count=1, user mode
  g0.y = ldsOff;                                     // lds_addr (bytes)
  g0.z = (unsigned)(ga & 0xFFFFFFFFu);               // global_addr[31:0]
  g0.w = (unsigned)((ga >> 32) & 0x01FFFFFFu)        // global_addr[56:32]
         | (2u << 30);                               // type = 2 ("image")
  v8i g1;
  g1[0] = (int)0x00020000u;                          // data_size = 4B
  g1[1] = (int)((PRE_TILE_ELEMS & 0xFFFF) << 16);    // tensor_dim0 lo16
  g1[2] = (int)(((PRE_TILE_ELEMS >> 16) & 0xFFFF)    // tensor_dim0 hi16
                | (1u << 16));                       // tensor_dim1 = 1 (lo16)
  g1[3] = (int)((PRE_TILE_ELEMS & 0xFFFF) << 16);    // tile_dim0 = 16384
  g1[4] = 1;                                         // tile_dim1 = 1
  g1[5] = PRE_TILE_ELEMS;                            // tensor_dim0_stride lo32
  g1[6] = 0;
  g1[7] = 0;
  v4i z4 = {0, 0, 0, 0};
  v8i z8 = {0, 0, 0, 0, 0, 0, 0, 0};
  __builtin_amdgcn_tensor_load_to_lds(g0, g1, z4, z4, z8, 0);
}

// ---------------------------------------------------------------------------
// Input-projection GEMM: C[M,N] f32 = A[M,K] f16 * Wp(packed)^T + b1 + b2
// Block = 8 waves; each wave owns one 16x16 tile; block covers 16(M) x 128(N).
// grid.x = (M/16) * (N/128)
// ---------------------------------------------------------------------------
__global__ __launch_bounds__(256)
void gemm_pre(const _Float16* __restrict__ A, const _Float16* __restrict__ Wp,
              const float* __restrict__ b1, const float* __restrict__ b2,
              float* __restrict__ C, int M, int N, int K) {
  const int lane = threadIdx.x & 31;
  const int wave = threadIdx.x >> 5;
  const int l15  = lane & 15;
  const int half = lane >> 4;
  const int ngroups = N >> 7;                 // groups of 8 n-tiles
  const int mt = blockIdx.x / ngroups;
  const int ng = blockIdx.x % ngroups;
  const int nt = ng * 8 + wave;
  const int m0 = mt * 16;
  const int n0 = nt * 16;
  const int KT = K >> 5;

  v8f acc = {};
  const _Float16* wp = Wp + ((size_t)nt * KT) * 512 + (size_t)lane * 16;
  const _Float16* arow = A + (size_t)(m0 + l15) * K;
  for (int kt = 0; kt < KT; ++kt) {
    v16h a = load_a_frag(arow + kt * 32, half);
    v16h b = *(const v16h*)wp;
    wp += 512;                                 // next k-tile fragment
    acc = __builtin_amdgcn_wmma_f32_16x16x32_f16(false, a, false, b,
                                                 (short)0, acc, false, false);
  }
  float bias = b1[n0 + l15] + (b2 ? b2[n0 + l15] : 0.0f);
#pragma unroll
  for (int r = 0; r < 8; ++r) {
    int m = m0 + r + 8 * half;                 // C layout: lane&15 = n, r(+8*half) = m
    C[(size_t)m * N + n0 + l15] = acc[r] + bias;
  }
}

// ---------------------------------------------------------------------------
// Persistent bidirectional LSTM scan (one layer, both directions).
// grid.x = 16: dir = blockIdx.x & 1, batch slice = blockIdx.x >> 1 (16 rows).
// Per step: gates[16,1024] = h[16,256] @ Whh^T (WMMA, 8 waves x 8 n-tiles
// x 8 k-tiles) + pre[t]; pre[t] slices are TDM-prefetched one step ahead
// into a 2x64KB dynamic-LDS ring. Then i,f,g,o -> c,h; h kept in LDS f16.
// Writes h (f16) into out[T,B,outLd] at feature offset dir*HID.
// ---------------------------------------------------------------------------
__global__ __launch_bounds__(256)
void lstm_scan(const float* __restrict__ preF, const float* __restrict__ preB,
               const _Float16* __restrict__ WpF, const _Float16* __restrict__ WpB,
               _Float16* __restrict__ out, int outLd) {
  __shared__ _Float16 hsh[16 * HID_];          // 8 KB: current h, f16
  __shared__ _Float16 gsh[16 * G4_];           // 32 KB: pre-activation gates
  extern __shared__ float preLds[];            // 2 x 16384 f32 = 128 KB ring

  const int dir = blockIdx.x & 1;
  const int b0  = (blockIdx.x >> 1) * 16;
  const float*    pre = dir ? preB : preF;
  const _Float16* Wp  = dir ? WpB  : WpF;
  const int off = dir ? HID_ : 0;

  const int lane = threadIdx.x & 31;
  const int wave = threadIdx.x >> 5;
  const int l15  = lane & 15;
  const int half = lane >> 4;
  const int j    = threadIdx.x;                // owned gate/hidden column 0..255

  const unsigned ldsBase = (unsigned)(uintptr_t)(void*)preLds; // LDS byte offset

  float c[16];
#pragma unroll
  for (int m = 0; m < 16; ++m) c[m] = 0.0f;
  for (int i = threadIdx.x; i < 16 * HID_; i += 256) hsh[i] = (_Float16)0.0f;

  // Prologue: kick off TDM for the first timestep's pre slice into buffer 0.
  if (wave == 0) {
    int t0 = dir ? (T_ - 1) : 0;
    tdm_load_pre(pre + ((size_t)t0 * B_ + b0) * G4_, ldsBase);
  }

  const int KT = HID_ >> 5;                    // 8 k-tiles
  for (int s = 0; s < T_; ++s) {
    const int t = dir ? (T_ - 1 - s) : s;

    // Prefetch step s+1 into the other ring slot; wait for slot s.
    if (wave == 0) {
      if (s + 1 < T_) {
        int tn = dir ? (T_ - 2 - s) : (s + 1);
        tdm_load_pre(pre + ((size_t)tn * B_ + b0) * G4_,
                     ldsBase + ((unsigned)((s + 1) & 1)) * (PRE_TILE_ELEMS * 4u));
        __builtin_amdgcn_s_wait_tensorcnt(1);  // oldest (slot s) complete
      } else {
        __builtin_amdgcn_s_wait_tensorcnt(0);
      }
    }
    __syncthreads();  // publishes: TDM pre[t] in slot s&1, h from step s-1

    const float* prow = preLds + (size_t)(s & 1) * PRE_TILE_ELEMS;

    // --- recurrent GEMM: wave owns gate columns [wave*128, wave*128+128) ---
    v8f acc[8];
#pragma unroll
    for (int q = 0; q < 8; ++q) acc[q] = (v8f){};
    for (int kt = 0; kt < KT; ++kt) {
      v16h a = load_a_frag(hsh + l15 * HID_ + kt * 32, half);
#pragma unroll
      for (int q = 0; q < 8; ++q) {
        int nt = wave * 8 + q;
        v16h b = *(const v16h*)(Wp + (((size_t)nt * KT + kt) * 32 + lane) * 16);
        acc[q] = __builtin_amdgcn_wmma_f32_16x16x32_f16(false, a, false, b,
                                                        (short)0, acc[q], false, false);
      }
    }
#pragma unroll
    for (int q = 0; q < 8; ++q) {
      int n = wave * 128 + q * 16 + l15;
#pragma unroll
      for (int r = 0; r < 8; ++r) {
        int m = r + 8 * half;
        gsh[m * G4_ + n] = (_Float16)(acc[q][r] + prow[m * G4_ + n]);
      }
    }
    __syncthreads();

    // --- elementwise cell update: thread j owns column j for all 16 rows ---
#pragma unroll 4
    for (int m = 0; m < 16; ++m) {
      float ig = (float)gsh[m * G4_ + j];
      float fg = (float)gsh[m * G4_ + HID_ + j];
      float gg = (float)gsh[m * G4_ + 2 * HID_ + j];
      float og = (float)gsh[m * G4_ + 3 * HID_ + j];
      float si = 1.0f / (1.0f + __expf(-ig));
      float sf = 1.0f / (1.0f + __expf(-fg));
      float so = 1.0f / (1.0f + __expf(-og));
      float tg = tanhf(gg);
      c[m] = sf * c[m] + si * tg;
      float h = so * tanhf(c[m]);
      hsh[m * HID_ + j] = (_Float16)h;
      out[((size_t)t * B_ + b0 + m) * outLd + off + j] = (_Float16)h;
    }
    // no trailing barrier: top-of-loop barrier covers hsh WAR/RAW
  }
}

// ---------------------------------------------------------------------------
// Output projection: tag[b][t][n] = A[t*B+b, 0:512] . w_out[n] + b_out[n]
// N padded to 64 in packed weights; block = 4 waves, one 16-row M tile.
// ---------------------------------------------------------------------------
__global__ __launch_bounds__(128)
void out_proj(const _Float16* __restrict__ A, const _Float16* __restrict__ Wp,
              const float* __restrict__ bias, float* __restrict__ outp) {
  const int lane = threadIdx.x & 31;
  const int wave = threadIdx.x >> 5;           // nt = wave (0..3 -> n 0..63)
  const int l15  = lane & 15;
  const int half = lane >> 4;
  const int m0   = blockIdx.x * 16;
  const int K = 2 * HID_, KT = K >> 5;         // 512, 16

  v8f acc = {};
  const _Float16* wp = Wp + ((size_t)wave * KT) * 512 + (size_t)lane * 16;
  const _Float16* arow = A + (size_t)(m0 + l15) * K;
  for (int kt = 0; kt < KT; ++kt) {
    v16h a = load_a_frag(arow + kt * 32, half);
    v16h b = *(const v16h*)wp;
    wp += 512;
    acc = __builtin_amdgcn_wmma_f32_16x16x32_f16(false, a, false, b,
                                                 (short)0, acc, false, false);
  }
  int n = wave * 16 + l15;
  if (n < TAGS_) {
    float bv = bias[n];
#pragma unroll
    for (int r = 0; r < 8; ++r) {
      int m  = m0 + r + 8 * half;              // m = t*B + b
      int bb = m & (B_ - 1);
      int tt = m >> 7;
      outp[((size_t)bb * T_ + tt) * TAGS_ + n] = acc[r] + bv;
    }
  }
}

// ---------------------------------------------------------------------------
extern "C" void kernel_launch(void* const* d_in, const int* in_sizes, int n_in,
                              void* d_out, int out_size, void* d_ws, size_t ws_size,
                              hipStream_t stream) {
  (void)in_sizes; (void)n_in; (void)out_size; (void)ws_size;
  const int*   sent  = (const int*)d_in[0];
  const float* emb   = (const float*)d_in[1];
  const float* wih1f = (const float*)d_in[2];
  const float* whh1f = (const float*)d_in[3];
  const float* bih1f = (const float*)d_in[4];
  const float* bhh1f = (const float*)d_in[5];
  const float* wih1b = (const float*)d_in[6];
  const float* whh1b = (const float*)d_in[7];
  const float* bih1b = (const float*)d_in[8];
  const float* bhh1b = (const float*)d_in[9];
  const float* wih2f = (const float*)d_in[10];
  const float* whh2f = (const float*)d_in[11];
  const float* bih2f = (const float*)d_in[12];
  const float* bhh2f = (const float*)d_in[13];
  const float* wih2b = (const float*)d_in[14];
  const float* whh2b = (const float*)d_in[15];
  const float* bih2b = (const float*)d_in[16];
  const float* bhh2b = (const float*)d_in[17];
  const float* w_out = (const float*)d_in[18];
  const float* b_out = (const float*)d_in[19];

  // ---- workspace carve-up (aligned 256B). Layer-2 pre reuses layer-1 pre. ----
  char* ws = (char*)d_ws;
  size_t off = 0;
  auto carve = [&](size_t bytes) -> void* {
    void* p = ws + off;
    off = (off + bytes + 255) & ~(size_t)255;
    return p;
  };
  _Float16* xs    = (_Float16*)carve((size_t)MROWS * EMB_ * 2);     // 8 MB
  _Float16* x2    = (_Float16*)carve((size_t)MROWS * 512 * 2);      // 32 MB
  _Float16* hcat  = (_Float16*)carve((size_t)MROWS * 512 * 2);      // 32 MB
  float*    preF  = (float*)carve((size_t)MROWS * G4_ * 4);         // 128 MB
  float*    preB  = (float*)carve((size_t)MROWS * G4_ * 4);         // 128 MB
  _Float16* p_wih1f = (_Float16*)carve((size_t)G4_ * EMB_ * 2);
  _Float16* p_wih1b = (_Float16*)carve((size_t)G4_ * EMB_ * 2);
  _Float16* p_whh1f = (_Float16*)carve((size_t)G4_ * HID_ * 2);
  _Float16* p_whh1b = (_Float16*)carve((size_t)G4_ * HID_ * 2);
  _Float16* p_wih2f = (_Float16*)carve((size_t)G4_ * 512 * 2);
  _Float16* p_wih2b = (_Float16*)carve((size_t)G4_ * 512 * 2);
  _Float16* p_whh2f = (_Float16*)carve((size_t)G4_ * HID_ * 2);
  _Float16* p_whh2b = (_Float16*)carve((size_t)G4_ * HID_ * 2);
  _Float16* p_wout  = (_Float16*)carve((size_t)64 * 512 * 2);

  auto pack = [&](const float* W, _Float16* P, int N, int Npad, int K) {
    int total = Npad * K;
    pack_w16<<<(total + 255) / 256, 256, 0, stream>>>(W, P, N, K, total);
  };
  pack(wih1f, p_wih1f, G4_, G4_, EMB_);
  pack(wih1b, p_wih1b, G4_, G4_, EMB_);
  pack(whh1f, p_whh1f, G4_, G4_, HID_);
  pack(whh1b, p_whh1b, G4_, G4_, HID_);
  pack(wih2f, p_wih2f, G4_, G4_, 512);
  pack(wih2b, p_wih2b, G4_, G4_, 512);
  pack(whh2f, p_whh2f, G4_, G4_, HID_);
  pack(whh2b, p_whh2b, G4_, G4_, HID_);
  pack(w_out, p_wout, TAGS_, 64, 512);

  embed_gather<<<(MROWS * EMB_ + 255) / 256, 256, 0, stream>>>(sent, emb, xs);

  // Allow 128KB dynamic LDS for the scan's TDM ring (static 40KB + 128KB).
  const int dynLds = 2 * PRE_TILE_ELEMS * (int)sizeof(float);
  (void)hipFuncSetAttribute(reinterpret_cast<const void*>(&lstm_scan),
                            hipFuncAttributeMaxDynamicSharedMemorySize, dynLds);

  const int gridPre = (MROWS / 16) * (G4_ / 128);                   // 16384
  // layer-1 input projections
  gemm_pre<<<gridPre, 256, 0, stream>>>(xs, p_wih1f, bih1f, bhh1f, preF,
                                        MROWS, G4_, EMB_);
  gemm_pre<<<gridPre, 256, 0, stream>>>(xs, p_wih1b, bih1b, bhh1b, preB,
                                        MROWS, G4_, EMB_);
  // layer-1 bidirectional scan -> x2 [T,B,512]
  lstm_scan<<<16, 256, dynLds, stream>>>(preF, preB, p_whh1f, p_whh1b, x2, 512);
  // layer-2 input projections (reuse pre buffers)
  gemm_pre<<<gridPre, 256, 0, stream>>>(x2, p_wih2f, bih2f, bhh2f, preF,
                                        MROWS, G4_, 512);
  gemm_pre<<<gridPre, 256, 0, stream>>>(x2, p_wih2b, bih2b, bhh2b, preB,
                                        MROWS, G4_, 512);
  // layer-2 bidirectional scan -> hcat [T,B,512]
  lstm_scan<<<16, 256, dynLds, stream>>>(preF, preB, p_whh2f, p_whh2b, hcat, 512);
  // output projection -> d_out [B,T,TAGS] f32
  out_proj<<<MROWS / 16, 128, 0, stream>>>(hcat, p_wout, b_out, (float*)d_out);
}
